// Model_36653250904671
// MI455X (gfx1250) — matrix operand order repaired and checked by reading, not compiled, not run on hardware
//
#include <hip/hip_runtime.h>
#include <cstdint>
#include <cstddef>

// ---------------------------------------------------------------------------
// 2-layer bidirectional LSTM LM for MI455X (gfx1250, wave32, WMMA bf16)
//   V=128 E=256 H=512 (4H=2048) B=128 T=512
// Strategy:
//   * bf16 V_WMMA_F32_16X16X32_BF16 everywhere (f32 accumulate)
//   * recurrence: 16 persistent workgroups (8 batch-tiles x 2 directions),
//     whole T-loop inside the kernel; h-state + gate preacts in LDS
//   * input projection fused into the per-step GEMM (K = In + H concat)
//   * Whh streamed from L2 (192MB L2 >> 18MB of bf16 weights)
// ---------------------------------------------------------------------------

typedef __bf16 bf16;
typedef __attribute__((ext_vector_type(16))) __bf16 v16bf;
typedef __attribute__((ext_vector_type(8)))  float  v8f;

#define V_  128
#define E_  256
#define H_  512
#define G_  2048   // 4*H
#define B_  128
#define T_  512
#define BT_ 16     // batch tile per workgroup

union AB16 { v16bf v; uint32_t u[8]; };

// 16-bit A-matrix 16x32 layout (ISA 7.12.2): lane = {half=lane>>4, m=lane&15};
// VGPR v holds K pair: v<4 -> K = half*8 + 2v ; v>=4 -> K = 16 + half*8 + 2(v-4)
__device__ inline int kA(int v, int half) {
  return ((v & 4) << 2) + (half << 3) + ((v & 3) << 1);
}
// B-matrix 32x16: same K mapping as A: v<4 -> K = half*8 + 2v ;
// v>=4 -> K = 16 + half*8 + 2(v-4).
__device__ inline int kB(int v, int half) {
  return ((v & 4) << 2) + (half << 3) + ((v & 3) << 1);
}

__device__ inline v8f wmma_bf16(v16bf a, v16bf b, v8f c) {
  // (neg_a, A, neg_b, B, c_mod, C, reuse_a, reuse_b)
  return __builtin_amdgcn_wmma_f32_16x16x32_bf16(false, a, false, b,
                                                 (short)0, c, false, false);
}

// A fragment: per-lane row pointer (row m = lane&15), works for global or LDS.
__device__ inline v16bf load_a(const bf16* row, int k0, int half) {
  AB16 f;
#pragma unroll
  for (int v = 0; v < 8; ++v)
    f.u[v] = *(const uint32_t*)(row + k0 + kA(v, half));
  return f.v;
}

// B fragment: per-lane weight-row pointer (row n = lane&15), W is [N][K] row major
// so B[k][n] = W[n][k].
__device__ inline v16bf load_b(const bf16* wrow, int k0, int half) {
  AB16 f;
#pragma unroll
  for (int v = 0; v < 8; ++v)
    f.u[v] = *(const uint32_t*)(wrow + k0 + kB(v, half));
  return f.v;
}

__device__ inline float sigf(float x) { return 1.0f / (1.0f + __expf(-x)); }
__device__ inline float tanh_fast(float x) {
  x = fminf(fmaxf(x, -15.0f), 15.0f);
  float e = __expf(2.0f * x);
  return (e - 1.0f) / (e + 1.0f);
}

// ---------------------------------------------------------------------------
__global__ void cvt_bf16_kernel(const float* __restrict__ src,
                                bf16* __restrict__ dst, int n) {
  int i = blockIdx.x * blockDim.x + threadIdx.x;
  if (i < n) dst[i] = (bf16)src[i];
}

// ---------------------------------------------------------------------------
// One bidirectional LSTM layer.
// grid = 16 blocks: blockIdx[0..7] = forward batch tiles, [8..15] = backward.
// block = 512 threads = 16 waves; wave w owns gate columns n in [w*128, w*128+128).
// LDS: gates[16][2048] f32 (128KB) + hbuf[16][512] bf16 (16KB) = 144KB of 320KB.
template <int IN_DIM, int LAYER>
__global__ __launch_bounds__(512)
void lstm_layer_kernel(const int*  __restrict__ sentence, // [B][T]   (layer 0)
                       const bf16* __restrict__ emb_bf,   // [V][E]   (layer 0)
                       const bf16* __restrict__ h_in,     // [T][B][2H] (layer 1)
                       const bf16* __restrict__ Wih,      // [2][4H][IN_DIM]
                       const bf16* __restrict__ Whh,      // [2][4H][H]
                       const float* __restrict__ bih,     // [2][4H]
                       const float* __restrict__ bhh,     // [2][4H]
                       bf16* __restrict__ h_out)          // [T][B][2H]
{
  extern __shared__ char smem[];
  float* gates = (float*)smem;                                   // [BT_][G_]
  bf16*  hbuf  = (bf16*)(smem + (size_t)BT_ * G_ * sizeof(float)); // [BT_][H_]

  const int tid  = threadIdx.x;
  const int lane = tid & 31;
  const int wave = tid >> 5;       // 0..15
  const int half = lane >> 4;
  const int ln   = lane & 15;

  const int dir = blockIdx.x >> 3; // 0 fwd, 1 bwd
  const int b0  = (blockIdx.x & 7) * BT_;

  const bf16*  WihD = Wih + (size_t)dir * G_ * IN_DIM;
  const bf16*  WhhD = Whh + (size_t)dir * G_ * H_;
  const float* bihD = bih + dir * G_;
  const float* bhhD = bhh + dir * G_;

  // zero h0
  for (int i = tid; i < BT_ * H_; i += 512) hbuf[i] = (bf16)0.0f;

  // cell state: thread owns (b = tid>>5, j = (tid&31)*16 + u), u in [0,16)
  const int ab = tid >> 5;
  const int j0 = (tid & 31) * 16;
  float creg[16];
#pragma unroll
  for (int u = 0; u < 16; ++u) creg[u] = 0.0f;

  __syncthreads();

  const int n_base = wave * 128;
  const v8f vzero = {0.f, 0.f, 0.f, 0.f, 0.f, 0.f, 0.f, 0.f};

  for (int s = 0; s < T_; ++s) {
    const int t = dir ? (T_ - 1 - s) : s;

    // per-lane A-row pointer for the input projection (row m = ln)
    const bf16* arow;
    if constexpr (LAYER == 0) {
      int idx = sentence[(b0 + ln) * T_ + t];
      arow = emb_bf + (size_t)idx * E_;
    } else {
      arow = h_in + ((size_t)t * B_ + b0 + ln) * (2 * H_);
    }

    v8f acc[8];
#pragma unroll
    for (int j = 0; j < 8; ++j) acc[j] = vzero;

    // ---- input projection part: gates += x_t @ Wih^T -------------------
#pragma unroll 1
    for (int k0 = 0; k0 < IN_DIM; k0 += 32) {
      v16bf a = load_a(arow, k0, half);
#pragma unroll
      for (int j = 0; j < 8; ++j) {
        const bf16* wrow = WihD + (size_t)(n_base + j * 16 + ln) * IN_DIM;
        v16bf b = load_b(wrow, k0, half);
        acc[j] = wmma_bf16(a, b, acc[j]);
      }
    }

    // ---- recurrent part: gates += h_{t-1} @ Whh^T (A from LDS) ---------
#pragma unroll 1
    for (int k0 = 0; k0 < H_; k0 += 32) {
      v16bf a = load_a(hbuf + ln * H_, k0, half);
#pragma unroll
      for (int j = 0; j < 8; ++j) {
        const bf16* wrow = WhhD + (size_t)(n_base + j * 16 + ln) * H_;
        __builtin_prefetch(wrow + k0 + 32, 0, 1); // global_prefetch_b8 (L2 stream)
        v16bf b = load_b(wrow, k0, half);
        acc[j] = wmma_bf16(a, b, acc[j]);
      }
    }

    // ---- epilogue: + bias, spill gate preacts to LDS -------------------
    // C/D layout: lane -> n = ln, VGPR v -> m = v + 8*half
#pragma unroll
    for (int j = 0; j < 8; ++j) {
      int n = n_base + j * 16 + ln;
      float bias = bihD[n] + bhhD[n];
#pragma unroll
      for (int v = 0; v < 8; ++v)
        gates[(v + half * 8) * G_ + n] = acc[j][v] + bias;
    }
    __syncthreads();

    // ---- activations: i,f,g,o -> c,h; write h to LDS + global ----------
    float* grow = gates + ab * G_;
    bf16* hg = h_out + ((size_t)t * B_ + b0 + ab) * (2 * H_) + dir * H_;
#pragma unroll
    for (int u = 0; u < 16; ++u) {
      int j = j0 + u;
      float gi = grow[j];
      float gf = grow[H_ + j];
      float gg = grow[2 * H_ + j];
      float go = grow[3 * H_ + j];
      float cn = sigf(gf) * creg[u] + sigf(gi) * tanh_fast(gg);
      creg[u] = cn;
      float hv = sigf(go) * tanh_fast(cn);
      bf16 hb = (bf16)hv;
      hbuf[ab * H_ + j] = hb;
      hg[j] = hb;
    }
    __syncthreads();
  }
}

// ---------------------------------------------------------------------------
// Output projection: out[b][t][v] = h1[t][b][:] @ Wout^T + bout
// 256 threads = 8 waves; each wave: 16 rows of (b*T+t) x all 128 vocab cols.
__global__ __launch_bounds__(256)
void out_proj_kernel(const bf16* __restrict__ h1,    // [T][B][2H]
                     const bf16* __restrict__ Wout,  // [V][2H]
                     const float* __restrict__ bout, // [V]
                     float* __restrict__ out)        // [B][T][V]
{
  const int tid  = threadIdx.x;
  const int lane = tid & 31;
  const int wave = tid >> 5; // 0..7
  const int half = lane >> 4;
  const int ln   = lane & 15;

  const int rbase = blockIdx.x * 128 + wave * 16; // rows over r = b*T + t
  int r = rbase + ln;
  int b = r >> 9;          // / T_
  int t = r & (T_ - 1);
  const bf16* arow = h1 + ((size_t)t * B_ + b) * (2 * H_);

  const v8f vzero = {0.f, 0.f, 0.f, 0.f, 0.f, 0.f, 0.f, 0.f};
  v8f acc[8];
#pragma unroll
  for (int j = 0; j < 8; ++j) acc[j] = vzero;

#pragma unroll 1
  for (int k0 = 0; k0 < 2 * H_; k0 += 32) {
    v16bf a = load_a(arow, k0, half);
#pragma unroll
    for (int j = 0; j < 8; ++j) {
      const bf16* wrow = Wout + (size_t)(j * 16 + ln) * (2 * H_);
      v16bf bb = load_b(wrow, k0, half);
      acc[j] = wmma_bf16(a, bb, acc[j]);
    }
  }

#pragma unroll
  for (int j = 0; j < 8; ++j) {
    int n = j * 16 + ln;
    float bv = bout[n];
#pragma unroll
    for (int v = 0; v < 8; ++v) {
      int m  = rbase + v + half * 8;
      int bb = m >> 9;
      int tt = m & (T_ - 1);
      out[((size_t)bb * T_ + tt) * V_ + n] = acc[j][v] + bv;
    }
  }
}

// ---------------------------------------------------------------------------
extern "C" void kernel_launch(void* const* d_in, const int* in_sizes, int n_in,
                              void* d_out, int out_size, void* d_ws, size_t ws_size,
                              hipStream_t stream) {
  const int*   sentence = (const int*)  d_in[0];
  const float* emb  = (const float*)d_in[1];
  const float* Wih0 = (const float*)d_in[2];
  const float* Whh0 = (const float*)d_in[3];
  const float* bih0 = (const float*)d_in[4];
  const float* bhh0 = (const float*)d_in[5];
  const float* Wih1 = (const float*)d_in[6];
  const float* Whh1 = (const float*)d_in[7];
  const float* bih1 = (const float*)d_in[8];
  const float* bhh1 = (const float*)d_in[9];
  const float* Wout = (const float*)d_in[10];
  const float* bout = (const float*)d_in[11];
  float* out = (float*)d_out;

  // workspace carve-up (~275 MB total)
  char*  ws  = (char*)d_ws;
  size_t off = 0;
  auto alloc = [&](size_t bytes) -> char* {
    char* p = ws + off;
    off = (off + bytes + 255) & ~(size_t)255;
    return p;
  };
  bf16* emb_bf  = (bf16*)alloc((size_t)V_ * E_ * 2);
  bf16* Wih0_bf = (bf16*)alloc((size_t)2 * G_ * E_ * 2);
  bf16* Whh0_bf = (bf16*)alloc((size_t)2 * G_ * H_ * 2);
  bf16* Wih1_bf = (bf16*)alloc((size_t)2 * G_ * 2 * H_ * 2);
  bf16* Whh1_bf = (bf16*)alloc((size_t)2 * G_ * H_ * 2);
  bf16* Wout_bf = (bf16*)alloc((size_t)V_ * 2 * H_ * 2);
  bf16* h0      = (bf16*)alloc((size_t)T_ * B_ * 2 * H_ * 2);
  bf16* h1      = (bf16*)alloc((size_t)T_ * B_ * 2 * H_ * 2);

  auto cvt = [&](const float* s, bf16* d, int n) {
    cvt_bf16_kernel<<<(n + 255) / 256, 256, 0, stream>>>(s, d, n);
  };
  cvt(emb,  emb_bf,  V_ * E_);
  cvt(Wih0, Wih0_bf, 2 * G_ * E_);
  cvt(Whh0, Whh0_bf, 2 * G_ * H_);
  cvt(Wih1, Wih1_bf, 2 * G_ * 2 * H_);
  cvt(Whh1, Whh1_bf, 2 * G_ * H_);
  cvt(Wout, Wout_bf, V_ * 2 * H_);

  const size_t smem = (size_t)BT_ * G_ * sizeof(float) + (size_t)BT_ * H_ * sizeof(bf16);

  lstm_layer_kernel<E_, 0><<<16, 512, smem, stream>>>(
      sentence, emb_bf, (const bf16*)nullptr, Wih0_bf, Whh0_bf, bih0, bhh0, h0);
  lstm_layer_kernel<2 * H_, 1><<<16, 512, smem, stream>>>(
      (const int*)nullptr, (const bf16*)nullptr, h0, Wih1_bf, Whh1_bf, bih1, bhh1, h1);
  out_proj_kernel<<<(B_ * T_) / 128, 256, 0, stream>>>(h1, Wout_bf, bout, out);
}
